// MambaCAN_6803228196992
// MI455X (gfx1250) — matrix operand order repaired and checked by reading, not compiled, not run on hardware
//
#include <hip/hip_runtime.h>
#include <hip/hip_bf16.h>
#include <math.h>

typedef __attribute__((ext_vector_type(16))) _Float16 v16h;
typedef __attribute__((ext_vector_type(8)))  _Float16 v8h;
typedef __attribute__((ext_vector_type(8)))  float    v8f;
typedef unsigned int uint32x4 __attribute__((ext_vector_type(4)));
typedef int          int32x8  __attribute__((ext_vector_type(8)));
typedef int          int32x4  __attribute__((ext_vector_type(4)));

#define BSZ     128
#define LSEQ    100
#define MROWS   (BSZ*LSEQ)     // 12800
#define DMODEL  512
#define DINNER  1024
#define DSTATE  16
#define DTRANK  32
#define HID     128
#define EMB     256

// block tile for the TDM-fed GEMM
#define BM 128
#define BN 128
#define BK 32

#if __has_builtin(__builtin_amdgcn_tensor_load_to_lds)
#define USE_TDM 1
#else
#define USE_TDM 0
#endif

// ---------------------------------------------------------------------------
// Generic fp32 -> fp16 convert
// ---------------------------------------------------------------------------
__global__ void k_f32_to_f16(const float* __restrict__ in, _Float16* __restrict__ out, int n) {
    int i = blockIdx.x * blockDim.x + threadIdx.x;
    if (i < n) out[i] = (_Float16)in[i];
}

__global__ void k_extract_dt16(const float* __restrict__ xdbl, _Float16* __restrict__ out, int n) {
    int i = blockIdx.x * blockDim.x + threadIdx.x;
    if (i < n) {
        int m = i / DTRANK, c = i % DTRANK;
        out[i] = (_Float16)xdbl[m * 64 + c];
    }
}

// ---------------------------------------------------------------------------
// Embedding + LayerNorm + ReLU  -> feat[:, 0:256]
// ---------------------------------------------------------------------------
__global__ __launch_bounds__(EMB) void k_embed_ln(
    const float* __restrict__ x, const float* __restrict__ emb_w,
    const float* __restrict__ emb_b, const float* __restrict__ ln_w,
    const float* __restrict__ ln_b, float* __restrict__ feat)
{
    __shared__ float red[EMB];
    const int m = blockIdx.x;
    const int b = m / LSEQ, t = m % LSEQ;
    const int c = threadIdx.x;
    const float val = x[b * 900 + t * 9 + 0];
    float e = val * emb_w[c] + emb_b[c];

    red[c] = e; __syncthreads();
    for (int s = EMB / 2; s > 0; s >>= 1) { if (c < s) red[c] += red[c + s]; __syncthreads(); }
    const float mean = red[0] / EMB; __syncthreads();
    float d = e - mean;
    red[c] = d * d; __syncthreads();
    for (int s = EMB / 2; s > 0; s >>= 1) { if (c < s) red[c] += red[c + s]; __syncthreads(); }
    const float var = red[0] / EMB;
    float o = d * rsqrtf(var + 1e-5f) * ln_w[c] + ln_b[c];
    feat[(size_t)m * DMODEL + c] = fmaxf(o, 0.f);
}

// ---------------------------------------------------------------------------
// conv1: (B,8,100) -> relu -> f1 (B,128,100), k=3 pad=1
// ---------------------------------------------------------------------------
__global__ __launch_bounds__(128) void k_conv1(
    const float* __restrict__ x, const float* __restrict__ w,
    const float* __restrict__ bias, float* __restrict__ f1)
{
    const int bo = blockIdx.x, b = bo / HID, o = bo % HID;
    const int t = threadIdx.x;
    if (t >= LSEQ) return;
    float acc = bias[o];
    for (int c = 0; c < 8; ++c)
        for (int kk = 0; kk < 3; ++kk) {
            int tt = t - 1 + kk;
            if (tt >= 0 && tt < LSEQ)
                acc += w[(o * 8 + c) * 3 + kk] * x[b * 900 + tt * 9 + 1 + c];
        }
    f1[((size_t)b * HID + o) * LSEQ + t] = fmaxf(acc, 0.f);
}

// conv2: (B,128,100) -> relu -> feat[:, 256+o], k=3 pad=1
__global__ __launch_bounds__(128) void k_conv2(
    const float* __restrict__ f1, const float* __restrict__ w,
    const float* __restrict__ bias, float* __restrict__ feat)
{
    const int bo = blockIdx.x, b = bo / 256, o = bo % 256;
    const int t = threadIdx.x;
    if (t >= LSEQ) return;
    float acc = bias[o];
    for (int c = 0; c < HID; ++c) {
        const float* wrow = w + (size_t)(o * HID + c) * 3;
        const float* frow = f1 + ((size_t)b * HID + c) * LSEQ;
        for (int kk = 0; kk < 3; ++kk) {
            int tt = t - 1 + kk;
            if (tt >= 0 && tt < LSEQ) acc += wrow[kk] * frow[tt];
        }
    }
    feat[((size_t)b * LSEQ + t) * DMODEL + 256 + o] = fmaxf(acc, 0.f);
}

// ---------------------------------------------------------------------------
// TDM: DMA a [rows x cols] f16 tile (row stride = strideElems) into LDS.
// Descriptor built per CDNA5 ISA D# layout (CDNA5 ISA).
// 6-arg builtin form: (group0 u32x4, group1 i32x8, group2 i32x4, group3 i32x4,
//                      extra i32x8, cpol) — groups 2/3/extra unused for 2D tiles.
// ---------------------------------------------------------------------------
#if USE_TDM
__device__ inline void tdm_load_tile_f16(unsigned ldsOff, const _Float16* gsrc,
                                         int rows, int cols, int strideElems)
{
    unsigned long long ga = (unsigned long long)(size_t)gsrc;
    uint32x4 g0;
    g0[0] = 1u;                                   // count=1, user mode
    g0[1] = ldsOff;                               // lds_addr (bytes)
    g0[2] = (unsigned)(ga & 0xFFFFFFFFu);         // global_addr[31:0]
    g0[3] = (unsigned)((ga >> 32) & 0x01FFFFFFu)  // global_addr[56:32]
          | 0x80000000u;                          // type = 2 ("image")
    int32x8 g1;
    g1[0] = 1 << 16;                              // wg_mask=0, data_size=1 (2B)
    g1[1] = (cols & 0xFFFF) << 16;                // tensor_dim0[15:0]
    g1[2] = ((cols >> 16) & 0xFFFF) | ((rows & 0xFFFF) << 16);  // td0 hi | td1 lo
    g1[3] = ((rows >> 16) & 0xFFFF) | ((cols & 0xFFFF) << 16);  // td1 hi | tile_dim0
    g1[4] = rows & 0xFFFF;                        // tile_dim1 (tile_dim2=0)
    g1[5] = strideElems;                          // tensor_dim0_stride[31:0]
    g1[6] = 0;                                    // stride0 hi | stride1 lo
    g1[7] = 0;
    int32x4 z4 = {0, 0, 0, 0};
    int32x8 z8 = {0, 0, 0, 0, 0, 0, 0, 0};
    __builtin_amdgcn_tensor_load_to_lds(g0, g1, z4, z4, z8, 0);
}
#endif

// Fallback synchronous tile copy: 256 threads, [rows x 32] f16 tile.
__device__ inline void sync_load_tile(_Float16* dst, const _Float16* gsrc,
                                      int strideElems)
{
    const int tid = threadIdx.x;
#pragma unroll
    for (int rep = 0; rep < 2; ++rep) {
        int c = tid + rep * 256;                  // chunk id, 512 chunks of 8 f16
        int row = c >> 2, col = (c & 3) * 8;
        *(v8h*)(dst + row * BK + col) = *(const v8h*)(gsrc + (size_t)row * strideElems + col);
    }
}

// ---------------------------------------------------------------------------
// Tiled WMMA GEMM with TDM-fed LDS double buffering.
// C[M,N] = A[M,K] * W[N,K]^T ; block tile 128x128, K step 32, 8 waves.
// Wave (wm,wn) computes rows [wm*32,+32) x cols [wn*64,+64): 2x4 wmma tiles.
// ---------------------------------------------------------------------------
__global__ __launch_bounds__(256) void k_gemm_wmma_tdm(
    const _Float16* __restrict__ A, const _Float16* __restrict__ W,
    float* __restrict__ C, int M, int N, int K)
{
    __shared__ __align__(16) _Float16 sA[2][BM * BK];
    __shared__ __align__(16) _Float16 sB[2][BN * BK];
    const int tid  = threadIdx.x;
    const int lane = tid & 31;
    const int wave = tid >> 5;
    const int wm   = wave >> 1;        // 0..3 -> M strip
    const int wn   = wave & 1;         // 0..1 -> N strip
    const int half = lane >> 4;
    const int l16  = lane & 15;
    const int nbn  = N / BN;
    const int bm   = blockIdx.x / nbn;
    const int bn   = blockIdx.x % nbn;
    const int nk   = K / BK;

    const _Float16* gA = A + (size_t)bm * BM * K;
    const _Float16* gW = W + (size_t)bn * BN * K;

    v8f acc[2][4];
#pragma unroll
    for (int mt = 0; mt < 2; ++mt)
#pragma unroll
        for (int nt = 0; nt < 4; ++nt) acc[mt][nt] = (v8f){};

#if USE_TDM
    if (tid < 32) {
        tdm_load_tile_f16((unsigned)(size_t)&sA[0][0], gA, BM, BK, K);
        tdm_load_tile_f16((unsigned)(size_t)&sB[0][0], gW, BN, BK, K);
    }
#else
    sync_load_tile(sA[0], gA, K);
    sync_load_tile(sB[0], gW, K);
#endif

    for (int kt = 0; kt < nk; ++kt) {
        const int buf = kt & 1;
#if USE_TDM
        if (tid < 32) {
            if (kt + 1 < nk) {
                tdm_load_tile_f16((unsigned)(size_t)&sA[buf ^ 1][0], gA + (kt + 1) * BK, BM, BK, K);
                tdm_load_tile_f16((unsigned)(size_t)&sB[buf ^ 1][0], gW + (kt + 1) * BK, BN, BK, K);
                __builtin_amdgcn_s_wait_tensorcnt((short)2);  // current buf landed
            } else {
                __builtin_amdgcn_s_wait_tensorcnt((short)0);
            }
        }
#endif
        __syncthreads();   // LDS tile visible to all waves

        // Build fragments from LDS (ds_load_b128 gathers per WMMA VGPR layout)
        v16h af[2], bf[4];
#pragma unroll
        for (int mt = 0; mt < 2; ++mt) {
            const int base = (wm * 32 + mt * 16 + l16) * BK;
            v8h a0 = *(const v8h*)&sA[buf][base + half * 16];
            v8h a1 = *(const v8h*)&sA[buf][base + half * 16 + 8];
#pragma unroll
            for (int i = 0; i < 8; ++i) { af[mt][i] = a0[i]; af[mt][8 + i] = a1[i]; }
        }
#pragma unroll
        for (int nt = 0; nt < 4; ++nt) {
            const int base = (wn * 64 + nt * 16 + l16) * BK;
            v8h b0 = *(const v8h*)&sB[buf][base + half * 16];
            v8h b1 = *(const v8h*)&sB[buf][base + half * 16 + 8];
#pragma unroll
            for (int i = 0; i < 8; ++i) { bf[nt][i] = b0[i]; bf[nt][8 + i] = b1[i]; }
        }
#pragma unroll
        for (int mt = 0; mt < 2; ++mt)
#pragma unroll
            for (int nt = 0; nt < 4; ++nt)
                acc[mt][nt] = __builtin_amdgcn_wmma_f32_16x16x32_f16(
                    false, af[mt], false, bf[nt], (short)0, acc[mt][nt], false, false);

        __syncthreads();   // all reads done before next overwrite
#if !USE_TDM
        if (kt + 1 < nk) {
            sync_load_tile(sA[buf ^ 1], gA + (kt + 1) * BK, K);
            sync_load_tile(sB[buf ^ 1], gW + (kt + 1) * BK, K);
        }
#endif
    }

#pragma unroll
    for (int mt = 0; mt < 2; ++mt)
#pragma unroll
        for (int nt = 0; nt < 4; ++nt) {
            float* crow = C + (size_t)(bm * BM + wm * 32 + mt * 16 + half * 8) * N
                            + bn * BN + wn * 64 + nt * 16 + l16;
#pragma unroll
            for (int r = 0; r < 8; ++r) crow[(size_t)r * N] = acc[mt][nt][r];
        }
}

// ---------------------------------------------------------------------------
// Direct-global WMMA GEMM (kept for the small N=64 x_proj GEMM)
// ---------------------------------------------------------------------------
__global__ __launch_bounds__(256) void k_gemm_wmma(
    const _Float16* __restrict__ A, const _Float16* __restrict__ W,
    float* __restrict__ C, int M, int N, int K)
{
    const int lane = threadIdx.x & 31;
    const int wave = threadIdx.x >> 5;
    const int tilesN = N >> 4;
    const int tile = blockIdx.x * 8 + wave;
    if (tile >= (M >> 4) * tilesN) return;
    const int mt = tile / tilesN, nt = tile % tilesN;
    const int half = lane >> 4;
    const int l16  = lane & 15;
    const size_t arow = (size_t)(mt * 16 + l16) * K;
    const size_t brow = (size_t)(nt * 16 + l16) * K;

    v8f acc = {};
    for (int k0 = 0; k0 < K; k0 += 32) {
        v8h a0 = *(const v8h*)(A + arow + k0 + half * 16);
        v8h a1 = *(const v8h*)(A + arow + k0 + half * 16 + 8);
        v8h b0 = *(const v8h*)(W + brow + k0 + half * 16);
        v8h b1 = *(const v8h*)(W + brow + k0 + half * 16 + 8);
        v16h av, bv;
#pragma unroll
        for (int i = 0; i < 8; ++i) {
            av[i] = a0[i]; av[8 + i] = a1[i];
            bv[i] = b0[i]; bv[8 + i] = b1[i];
        }
        acc = __builtin_amdgcn_wmma_f32_16x16x32_f16(
            false, av, false, bv, (short)0, acc, false, false);
    }
    float* crow = C + (size_t)(mt * 16 + half * 8) * N + nt * 16 + l16;
#pragma unroll
    for (int r = 0; r < 8; ++r) crow[(size_t)r * N] = acc[r];
}

// ---------------------------------------------------------------------------
// Depthwise causal conv (k=4, left pad 3) + SiLU on u = xz[:, :1024]
// ---------------------------------------------------------------------------
__global__ void k_dwconv_silu(
    const float* __restrict__ xz, const float* __restrict__ w,
    const float* __restrict__ bias, float* __restrict__ u)
{
    int i = blockIdx.x * blockDim.x + threadIdx.x;
    if (i >= MROWS * DINNER) return;
    const int m = i / DINNER, d = i % DINNER;
    const int b = m / LSEQ, t = m % LSEQ;
    float acc = bias[d];
#pragma unroll
    for (int k = 0; k < 4; ++k) {
        int tt = t + k - 3;
        if (tt >= 0)
            acc += w[d * 4 + k] * xz[(size_t)(b * LSEQ + tt) * (2 * DINNER) + d];
    }
    u[(size_t)m * DINNER + d] = acc / (1.f + expf(-acc));
}

__global__ void k_dt_softplus(float* __restrict__ dt, const float* __restrict__ bias) {
    int i = blockIdx.x * blockDim.x + threadIdx.x;
    if (i >= MROWS * DINNER) return;
    float v = dt[i] + bias[i % DINNER];
    dt[i] = (v > 20.f) ? v : log1pf(expf(v));
}

__global__ void k_a_neg(const float* __restrict__ a_log, float* __restrict__ a_neg, int n) {
    int i = blockIdx.x * blockDim.x + threadIdx.x;
    if (i < n) a_neg[i] = -expf(a_log[i]);
}

// ---------------------------------------------------------------------------
// Selective scan + fused gating epilogue
// ---------------------------------------------------------------------------
__global__ __launch_bounds__(512) void k_scan(
    const float* __restrict__ dt, const float* __restrict__ u,
    const float* __restrict__ xdbl, const float* __restrict__ a_neg,
    const float* __restrict__ Dp, const float* __restrict__ xz,
    float* __restrict__ y)
{
    __shared__ float sB[DSTATE];
    __shared__ float sC[DSTATE];
    const int b = blockIdx.x;
    const int tid = threadIdx.x;
    const int d0 = tid * 2;

    float h[2][DSTATE];
    float A[2][DSTATE];
#pragma unroll
    for (int ch = 0; ch < 2; ++ch)
#pragma unroll
        for (int n = 0; n < DSTATE; ++n) {
            h[ch][n] = 0.f;
            A[ch][n] = a_neg[(d0 + ch) * DSTATE + n];
        }

    for (int t = 0; t < LSEQ; ++t) {
        const int m = b * LSEQ + t;
        if (tid < DSTATE)            sB[tid]      = xdbl[(size_t)m * 64 + 32 + tid];
        else if (tid < 2 * DSTATE)   sC[tid - 16] = xdbl[(size_t)m * 64 + 48 + (tid - 16)];
        __syncthreads();
#pragma unroll
        for (int ch = 0; ch < 2; ++ch) {
            const int d = d0 + ch;
            const float dt_t = dt[(size_t)m * DINNER + d];
            const float u_t  = u[(size_t)m * DINNER + d];
            const float du   = dt_t * u_t;
            float yacc = 0.f;
#pragma unroll
            for (int n = 0; n < DSTATE; ++n) {
                h[ch][n] = expf(dt_t * A[ch][n]) * h[ch][n] + du * sB[n];
                yacc += h[ch][n] * sC[n];
            }
            const float z = xz[(size_t)m * (2 * DINNER) + DINNER + d];
            const float silu_z = z / (1.f + expf(-z));
            y[(size_t)m * DINNER + d] = (yacc + u_t * Dp[d]) * silu_z;
        }
        __syncthreads();
    }
}

__global__ void k_maxpool(const float* __restrict__ attn, float* __restrict__ pooled) {
    int i = blockIdx.x * blockDim.x + threadIdx.x;
    if (i >= BSZ * DMODEL) return;
    const int b = i / DMODEL, j = i % DMODEL;
    float mx = -INFINITY;
    for (int t = 0; t < LSEQ; ++t)
        mx = fmaxf(mx, attn[(size_t)(b * LSEQ + t) * DMODEL + j]);
    pooled[i] = mx;
}

// ---------------------------------------------------------------------------
// Head: LN(512) -> fc1(512->128)+relu -> fc2(128->4)
// ---------------------------------------------------------------------------
__global__ __launch_bounds__(512) void k_head(
    const float* __restrict__ pooled,
    const float* __restrict__ nw, const float* __restrict__ nb,
    const float* __restrict__ w1, const float* __restrict__ b1,
    const float* __restrict__ w2, const float* __restrict__ b2,
    float* __restrict__ out)
{
    __shared__ float red[DMODEL];
    __shared__ float s[DMODEL];
    __shared__ float h1s[HID];
    const int b = blockIdx.x, c = threadIdx.x;
    const float v = pooled[(size_t)b * DMODEL + c];

    red[c] = v; __syncthreads();
    for (int st = DMODEL / 2; st > 0; st >>= 1) { if (c < st) red[c] += red[c + st]; __syncthreads(); }
    const float mean = red[0] / DMODEL; __syncthreads();
    const float d = v - mean;
    red[c] = d * d; __syncthreads();
    for (int st = DMODEL / 2; st > 0; st >>= 1) { if (c < st) red[c] += red[c + st]; __syncthreads(); }
    const float var = red[0] / DMODEL;
    s[c] = d * rsqrtf(var + 1e-5f) * nw[c] + nb[c];
    __syncthreads();

    if (c < HID) {
        float acc = b1[c];
        for (int j = 0; j < DMODEL; ++j) acc += w1[(size_t)c * DMODEL + j] * s[j];
        h1s[c] = fmaxf(acc, 0.f);
    }
    __syncthreads();
    if (c < 4) {
        float acc = b2[c];
        for (int i = 0; i < HID; ++i) acc += w2[c * HID + i] * h1s[i];
        out[b * 4 + c] = acc;
    }
}

// ---------------------------------------------------------------------------
extern "C" void kernel_launch(void* const* d_in, const int* in_sizes, int n_in,
                              void* d_out, int out_size, void* d_ws, size_t ws_size,
                              hipStream_t stream) {
    const float* x        = (const float*)d_in[0];
    const float* emb_w    = (const float*)d_in[1];
    const float* emb_b    = (const float*)d_in[2];
    const float* emb_lnw  = (const float*)d_in[3];
    const float* emb_lnb  = (const float*)d_in[4];
    const float* conv1_w  = (const float*)d_in[5];
    const float* conv1_b  = (const float*)d_in[6];
    const float* conv2_w  = (const float*)d_in[7];
    const float* conv2_b  = (const float*)d_in[8];
    const float* in_proj  = (const float*)d_in[9];
    const float* convd_w  = (const float*)d_in[10];
    const float* convd_b  = (const float*)d_in[11];
    const float* x_proj   = (const float*)d_in[12];
    const float* dt_proj  = (const float*)d_in[13];
    const float* dt_pb    = (const float*)d_in[14];
    const float* A_log    = (const float*)d_in[15];
    const float* Dp       = (const float*)d_in[16];
    const float* out_proj = (const float*)d_in[17];
    const float* norm_w   = (const float*)d_in[18];
    const float* norm_b   = (const float*)d_in[19];
    const float* fc1_w    = (const float*)d_in[20];
    const float* fc1_b    = (const float*)d_in[21];
    const float* fc2_w    = (const float*)d_in[22];
    const float* fc2_b    = (const float*)d_in[23];
    float* out = (float*)d_out;

    char* base = (char*)d_ws;
    size_t off = 0;
    auto alloc = [&](size_t bytes) -> char* {
        char* p = base + off;
        off += (bytes + 255) & ~(size_t)255;
        return p;
    };
    float*    f1       = (float*)   alloc((size_t)BSZ * HID * LSEQ * 4);
    float*    feat     = (float*)   alloc((size_t)MROWS * DMODEL * 4);
    _Float16* feat16   = (_Float16*)alloc((size_t)MROWS * DMODEL * 2);
    _Float16* inp16    = (_Float16*)alloc((size_t)2 * DINNER * DMODEL * 2);
    _Float16* xp16     = (_Float16*)alloc((size_t)64 * DINNER * 2);
    _Float16* dtp16    = (_Float16*)alloc((size_t)DINNER * DTRANK * 2);
    _Float16* outp16   = (_Float16*)alloc((size_t)DMODEL * DINNER * 2);
    float*    xz       = (float*)   alloc((size_t)MROWS * 2 * DINNER * 4);
    float*    u        = (float*)   alloc((size_t)MROWS * DINNER * 4);
    _Float16* u16      = (_Float16*)alloc((size_t)MROWS * DINNER * 2);
    float*    xdbl     = (float*)   alloc((size_t)MROWS * 64 * 4);
    _Float16* dt16     = (_Float16*)alloc((size_t)MROWS * DTRANK * 2);
    float*    dt       = (float*)   alloc((size_t)MROWS * DINNER * 4);
    float*    a_neg    = (float*)   alloc((size_t)DINNER * DSTATE * 4);
    float*    y        = (float*)   alloc((size_t)MROWS * DINNER * 4);
    _Float16* y16      = (_Float16*)alloc((size_t)MROWS * DINNER * 2);
    float*    attn     = (float*)   alloc((size_t)MROWS * DMODEL * 4);
    float*    pooled   = (float*)   alloc((size_t)BSZ * DMODEL * 4);
    (void)ws_size; (void)in_sizes; (void)n_in; (void)out_size;

    auto cvt = [&](const float* src, _Float16* dst, int n) {
        k_f32_to_f16<<<(n + 255) / 256, 256, 0, stream>>>(src, dst, n);
    };
    cvt(in_proj,  inp16,  2 * DINNER * DMODEL);
    cvt(x_proj,   xp16,   64 * DINNER);
    cvt(dt_proj,  dtp16,  DINNER * DTRANK);
    cvt(out_proj, outp16, DMODEL * DINNER);

    // ----- front end -----
    k_embed_ln<<<MROWS, EMB, 0, stream>>>(x, emb_w, emb_b, emb_lnw, emb_lnb, feat);
    k_conv1<<<BSZ * HID, 128, 0, stream>>>(x, conv1_w, conv1_b, f1);
    k_conv2<<<BSZ * 256, 128, 0, stream>>>(f1, conv2_w, conv2_b, feat);
    cvt(feat, feat16, MROWS * DMODEL);

    // ----- in_proj GEMM: [12800,512] x [2048,512]^T (TDM + LDS + WMMA) -----
    k_gemm_wmma_tdm<<<(MROWS / BM) * (2 * DINNER / BN), 256, 0, stream>>>(
        feat16, inp16, xz, MROWS, 2 * DINNER, DMODEL);

    // ----- depthwise conv + silu -----
    k_dwconv_silu<<<(MROWS * DINNER + 255) / 256, 256, 0, stream>>>(xz, convd_w, convd_b, u);
    cvt(u, u16, MROWS * DINNER);

    // ----- x_proj GEMM: [12800,1024] x [64,1024]^T (direct WMMA, N too small for 128-tile) -----
    {
        int tiles = (MROWS / 16) * (64 / 16);
        k_gemm_wmma<<<(tiles + 7) / 8, 256, 0, stream>>>(u16, xp16, xdbl, MROWS, 64, DINNER);
    }

    // ----- dt_proj GEMM: [12800,32] x [1024,32]^T, then softplus -----
    k_extract_dt16<<<(MROWS * DTRANK + 255) / 256, 256, 0, stream>>>(xdbl, dt16, MROWS * DTRANK);
    k_gemm_wmma_tdm<<<(MROWS / BM) * (DINNER / BN), 256, 0, stream>>>(
        dt16, dtp16, dt, MROWS, DINNER, DTRANK);
    k_dt_softplus<<<(MROWS * DINNER + 255) / 256, 256, 0, stream>>>(dt, dt_pb);

    // ----- selective scan -----
    k_a_neg<<<(DINNER * DSTATE + 255) / 256, 256, 0, stream>>>(A_log, a_neg, DINNER * DSTATE);
    k_scan<<<BSZ, 512, 0, stream>>>(dt, u, xdbl, a_neg, Dp, xz, y);

    // ----- out_proj GEMM: [12800,1024] x [512,1024]^T -----
    cvt(y, y16, MROWS * DINNER);
    k_gemm_wmma_tdm<<<(MROWS / BM) * (DMODEL / BN), 256, 0, stream>>>(
        y16, outp16, attn, MROWS, DMODEL, DINNER);

    // ----- pool + head -----
    k_maxpool<<<(BSZ * DMODEL + 255) / 256, 256, 0, stream>>>(attn, pooled);
    k_head<<<BSZ, DMODEL, 0, stream>>>(pooled, norm_w, norm_b, fc1_w, fc1_b, fc2_w, fc2_b, out);
}
